// GINLayer_7035156431318
// MI455X (gfx1250) — compile-verified, matrix operand order checked by reading, not run on hardware
//
#include <hip/hip_runtime.h>

#define N_NODES 100000
#define DIM     128
#define DIM2    256
#define N_EDGES 1600000
#define BN_EPS  1e-5f
#define NTILES  (N_NODES / 16)   // 6250, exact

typedef __attribute__((ext_vector_type(16))) __bf16   v16bf;
typedef __attribute__((ext_vector_type(2)))  __bf16   v2bf;
typedef __attribute__((ext_vector_type(8)))  float    v8f;
typedef __attribute__((ext_vector_type(8)))  unsigned v8u;

// ---- bf16 conversion via native HW converts (RNE) -------------------------
__device__ __forceinline__ unsigned pk2(float lo, float hi) {
  v2bf p;
  p[0] = (__bf16)lo;
  p[1] = (__bf16)hi;
  return __builtin_bit_cast(unsigned, p);
}
__device__ __forceinline__ unsigned short bf1(float f) {
  __bf16 b = (__bf16)f;
  return __builtin_bit_cast(unsigned short, b);
}

// K index inside a 16x32 bf16 A/B fragment (ISA 7.12.2 16-bit A-matrix layout):
// VGPR v (0..7), even pair element, lane half (0..1)
__device__ __forceinline__ int kk(int v, int half) {
  return ((v < 4) ? 0 : 16) + half * 8 + (v & 3) * 2;
}

// ---- h0 = (1 + eps) * x ---------------------------------------------------
__global__ void init_h0(const float* __restrict__ x, const float* __restrict__ eps,
                        float* __restrict__ h0, int n4) {
  int i = blockIdx.x * blockDim.x + threadIdx.x;
  if (i >= n4) return;
  float s = 1.0f + eps[0];
  float4 v = ((const float4*)x)[i];
  v.x *= s; v.y *= s; v.z *= s; v.w *= s;
  ((float4*)h0)[i] = v;
}

// ---- h0[dst] += x[src]  (one wave per edge, fp32 HW atomics) --------------
__global__ void scatter_edges(const float* __restrict__ x, const int* __restrict__ ei,
                              float* __restrict__ h0, int nE) {
  int e    = (blockIdx.x * blockDim.x + threadIdx.x) >> 5;
  int lane = threadIdx.x & 31;
  if (e >= nE) return;
  int src = ei[e];
  int dst = ei[nE + e];
  const float4 v = *(const float4*)(x + (size_t)src * DIM + lane * 4);
  float* p = h0 + (size_t)dst * DIM + lane * 4;
  unsafeAtomicAdd(p + 0, v.x);
  unsafeAtomicAdd(p + 1, v.y);
  unsafeAtomicAdd(p + 2, v.z);
  unsafeAtomicAdd(p + 3, v.w);
}

// ---- pre-swizzle W1/W2 into bf16 B-fragments (1KB each, lane-contiguous) ---
// frag 0..63   : W1, frag = ntile*4 + ktile   (K=128, N=256)
// frag 64..127 : W2, frag = 64 + ntile*8 + kt (K=256, N=128)
__global__ void prep_weights(const float* __restrict__ W1, const float* __restrict__ W2,
                             unsigned* __restrict__ wfrag) {
  int wave = (blockIdx.x * blockDim.x + threadIdx.x) >> 5;
  int lane = threadIdx.x & 31;
  if (wave >= 128) return;
  const float* W; int ncols, ntile, ktile;
  if (wave < 64) { W = W1; ncols = DIM2; ntile = wave >> 2;        ktile = wave & 3; }
  else           { W = W2; ncols = DIM;  ntile = (wave - 64) >> 3; ktile = (wave - 64) & 7; }
  int nloc = lane & 15, half = lane >> 4;
  int n = ntile * 16 + nloc;
  v8u t;
#pragma unroll
  for (int v = 0; v < 8; ++v) {
    int k0 = ktile * 32 + kk(v, half);
    t[v] = pk2(W[(size_t)k0 * ncols + n], W[(size_t)(k0 + 1) * ncols + n]);
  }
  *(v8u*)(wfrag + (size_t)wave * 256 + lane * 8) = t;   // 32B per lane
}

// ---- fused MLP: GEMM1 -> BN/ReLU -> GEMM2 -> BN/ReLU ----------------------
__global__ void __launch_bounds__(128)
gin_mlp(const float* __restrict__ h0, const unsigned* __restrict__ wfrag,
        const float* __restrict__ b1, const float* __restrict__ g1,
        const float* __restrict__ be1, const float* __restrict__ m1,
        const float* __restrict__ v1,
        const float* __restrict__ b2, const float* __restrict__ g2,
        const float* __restrict__ be2, const float* __restrict__ m2,
        const float* __restrict__ v2,
        float* __restrict__ out) {
  __shared__ unsigned short h1s[4][16][DIM2];   // 32KB: per-wave bf16 stash
  int wave = threadIdx.x >> 5;
  int lane = threadIdx.x & 31;
  int tile = blockIdx.x * 4 + wave;
  if (tile >= NTILES) return;
  int nloc = lane & 15, half = lane >> 4;

  const v16bf* wf = (const v16bf*)wfrag;        // fragment i : wf[i*32 + lane]

  // A fragments for GEMM1: 16 rows x 128 fp32 -> bf16, 4 K-tiles
  v16bf A1[4];
  const float* rowp = h0 + (size_t)(tile * 16 + nloc) * DIM;
#pragma unroll
  for (int kt = 0; kt < 4; ++kt) {
    v8u t;
#pragma unroll
    for (int v = 0; v < 8; ++v) {
      int k0 = kt * 32 + kk(v, half);
      float2 f = *(const float2*)(rowp + k0);
      t[v] = pk2(f.x, f.y);
    }
    A1[kt] = __builtin_bit_cast(v16bf, t);
  }

  // GEMM1: [16x128] @ [128x256], epilogue BN1+ReLU, stash bf16 to LDS
#pragma unroll
  for (int nt = 0; nt < 16; ++nt) {
    v8f acc = {};
#pragma unroll
    for (int kt = 0; kt < 4; ++kt) {
      v16bf B = wf[(nt * 4 + kt) * 32 + lane];
      acc = __builtin_amdgcn_wmma_f32_16x16x32_bf16(false, A1[kt], false, B,
                                                    (short)0, acc, false, false);
    }
    int col = nt * 16 + nloc;
    float sc = g1[col] * rsqrtf(v1[col] + BN_EPS);
    float sh = be1[col] + (b1[col] - m1[col]) * sc;
#pragma unroll
    for (int r = 0; r < 8; ++r) {
      float y = fmaxf(acc[r] * sc + sh, 0.0f);
      h1s[wave][r + 8 * half][col] = bf1(y);
    }
  }

  __builtin_amdgcn_wave_barrier();   // order LDS stash vs re-fragment loads

  // A fragments for GEMM2 from LDS (bf16 pairs, k0 always even -> dword reads)
  v16bf A2[8];
  const unsigned* h1w = (const unsigned*)&h1s[wave][0][0];
#pragma unroll
  for (int kt = 0; kt < 8; ++kt) {
    v8u t;
#pragma unroll
    for (int v = 0; v < 8; ++v) {
      int k0 = kt * 32 + kk(v, half);
      t[v] = h1w[(nloc * DIM2 + k0) >> 1];
    }
    A2[kt] = __builtin_bit_cast(v16bf, t);
  }

  // GEMM2: [16x256] @ [256x128], epilogue BN2+ReLU, fp32 out
#pragma unroll
  for (int nt = 0; nt < 8; ++nt) {
    v8f acc = {};
#pragma unroll
    for (int kt = 0; kt < 8; ++kt) {
      v16bf B = wf[(64 + nt * 8 + kt) * 32 + lane];
      acc = __builtin_amdgcn_wmma_f32_16x16x32_bf16(false, A2[kt], false, B,
                                                    (short)0, acc, false, false);
    }
    int col = nt * 16 + nloc;
    float sc = g2[col] * rsqrtf(v2[col] + BN_EPS);
    float sh = be2[col] + (b2[col] - m2[col]) * sc;
#pragma unroll
    for (int r = 0; r < 8; ++r) {
      float y = fmaxf(acc[r] * sc + sh, 0.0f);
      out[(size_t)(tile * 16 + r + 8 * half) * DIM + col] = y;
    }
  }
}

extern "C" void kernel_launch(void* const* d_in, const int* in_sizes, int n_in,
                              void* d_out, int out_size, void* d_ws, size_t ws_size,
                              hipStream_t stream) {
  const float* x   = (const float*)d_in[0];
  const float* eps = (const float*)d_in[1];
  const float* W1  = (const float*)d_in[2];
  const float* b1  = (const float*)d_in[3];
  const float* g1  = (const float*)d_in[4];
  const float* be1 = (const float*)d_in[5];
  const float* m1  = (const float*)d_in[6];
  const float* v1  = (const float*)d_in[7];
  const float* W2  = (const float*)d_in[8];
  const float* b2  = (const float*)d_in[9];
  const float* g2  = (const float*)d_in[10];
  const float* be2 = (const float*)d_in[11];
  const float* m2  = (const float*)d_in[12];
  const float* v2  = (const float*)d_in[13];
  const int*   ei  = (const int*)d_in[14];
  float* out = (float*)d_out;
  (void)in_sizes; (void)n_in; (void)out_size; (void)ws_size;

  float*    h0    = (float*)d_ws;                                       // 51.2 MB
  unsigned* wfrag = (unsigned*)((char*)d_ws +
                                (size_t)N_NODES * DIM * sizeof(float)); // 128 KB

  int n4 = N_NODES * DIM / 4;
  init_h0<<<(n4 + 255) / 256, 256, 0, stream>>>(x, eps, h0, n4);
  prep_weights<<<16, 256, 0, stream>>>(W1, W2, wfrag);
  scatter_edges<<<(N_EDGES + 7) / 8, 256, 0, stream>>>(x, ei, h0, N_EDGES);
  gin_mlp<<<(NTILES + 3) / 4, 128, 0, stream>>>(h0, wfrag,
                                                b1, g1, be1, m1, v1,
                                                b2, g2, be2, m2, v2, out);
}
